// Experts_3453153706553
// MI455X (gfx1250) — matrix operand order repaired and checked by reading, not compiled, not run on hardware
//
#include <hip/hip_runtime.h>
#include <hip/hip_bf16.h>
#include <math.h>

// Problem sizes (match reference)
#define NEXP 8
#define TOK  2048
#define HID  1024
#define INTR 4096

// GEMM tiling
#define BM 128
#define BN 128
#define BK 64          // two WMMA K-steps per LDS stage
#define LDST 36        // LDS row stride in dwords (32 k-pair dwords + pad), mult of 4

#define AS1 __attribute__((address_space(1)))
#define AS3 __attribute__((address_space(3)))

#if defined(__gfx1250__) && __has_builtin(__builtin_amdgcn_global_load_async_to_lds_b128)
#define USE_ASYNC_LDS 1
#else
#define USE_ASYNC_LDS 0
#endif

typedef __attribute__((ext_vector_type(16))) __bf16 v16bf;
typedef __attribute__((ext_vector_type(8)))  float  v8f;
typedef int vi4 __attribute__((vector_size(16)));   // native vector (matches builtin param)

union FragU { v16bf v; int4 i[2]; };

template <int N>
__device__ __forceinline__ void wait_asynccnt() {
#if __has_builtin(__builtin_amdgcn_s_wait_asynccnt)
  __builtin_amdgcn_s_wait_asynccnt(N);
#elif defined(__gfx1250__)
  asm volatile("s_wait_asynccnt %0" ::"i"(N) : "memory");
#endif
}

__device__ __forceinline__ unsigned bfpack(float lo, float hi) {
  unsigned ul = __float_as_uint(lo);
  unsigned uh = __float_as_uint(hi);
  ul = (ul + 0x7FFFu + ((ul >> 16) & 1u)) >> 16;   // RNE fp32 -> bf16
  uh = (uh + 0x7FFFu + ((uh >> 16) & 1u)) >> 16;
  return (uh << 16) | (ul & 0xFFFFu);
}

__device__ __forceinline__ unsigned short bf1(float x) {
  unsigned u = __float_as_uint(x);
  u = (u + 0x7FFFu + ((u >> 16) & 1u)) >> 16;
  return (unsigned short)u;
}

// ---------------------------------------------------------------------------
// X (fp32 row-major) -> bf16 row-major. Dwords along rows are K-pairs.
// ---------------------------------------------------------------------------
__global__ __launch_bounds__(256) void convert_x_kernel(
    const float* __restrict__ X, unsigned* __restrict__ Xb) {
  const size_t i = ((size_t)blockIdx.x * 256 + threadIdx.x) * 8;
  float4 a = ((const float4*)(X + i))[0];
  float4 b = ((const float4*)(X + i))[1];
  int4 o;
  o.x = (int)bfpack(a.x, a.y);
  o.y = (int)bfpack(a.z, a.w);
  o.z = (int)bfpack(b.x, b.y);
  o.w = (int)bfpack(b.z, b.w);
  *(int4*)(Xb + i / 2) = o;
}

// ---------------------------------------------------------------------------
// W (fp32, [K][N] per expert) -> WP (bf16 pair-dwords, [N][K/2] per expert).
// ---------------------------------------------------------------------------
__global__ __launch_bounds__(256) void convert_transpose_w_kernel(
    const float* __restrict__ W, unsigned* __restrict__ WP,
    int K, int N) {
  __shared__ float tile[64][65];
  const int e  = blockIdx.z;
  const int n0 = blockIdx.x * 64;
  const int k0 = blockIdx.y * 64;
  const float* We = W  + (size_t)e * K * N;
  unsigned*    Pe = WP + (size_t)e * N * (K / 2);

  const int t = threadIdx.x;
  {
    const int rb = t >> 4;
    const int c  = (t & 15) * 4;
    #pragma unroll
    for (int i = 0; i < 4; ++i) {
      const int r = rb + 16 * i;
      float4 v = *(const float4*)(We + (size_t)(k0 + r) * N + n0 + c);
      tile[r][c + 0] = v.x; tile[r][c + 1] = v.y;
      tile[r][c + 2] = v.z; tile[r][c + 3] = v.w;
    }
  }
  __syncthreads();
  {
    const int nl  = t >> 2;
    const int kp0 = (t & 3) * 8;
    unsigned* dst = Pe + (size_t)(n0 + nl) * (K / 2) + k0 / 2 + kp0;
    #pragma unroll
    for (int j = 0; j < 8; ++j) {
      const int kp = kp0 + j;
      dst[j] = bfpack(tile[2 * kp][nl], tile[2 * kp + 1][nl]);
    }
  }
}

// ---------------------------------------------------------------------------
// bf16 GEMM:  C[e] (TOK x NOUT) = A[e] (TOK x KTOT, bf16 row-major)
//                               @ B[e] ([NOUT][KTOT/2] pair-dwords)
// ---------------------------------------------------------------------------
template <int KTOT, int NOUT, bool DO_GELU>
__global__ __launch_bounds__(256, 1) void gemm_bf16_kernel(
    const unsigned short* __restrict__ A, const unsigned* __restrict__ BP,
    void* __restrict__ Cout) {
  __shared__ unsigned As[2][BM][LDST];   // [row][k-pair dword]
  __shared__ unsigned Bs[2][BN][LDST];   // [col][k-pair dword]

  const int e  = blockIdx.z;
  const int m0 = blockIdx.y * BM;
  const int n0 = blockIdx.x * BN;
  const unsigned short* Ae = A  + (size_t)e * TOK * KTOT;
  const unsigned*       Be = BP + (size_t)e * NOUT * (KTOT / 2);

  const int tid  = threadIdx.x;
  const int lane = tid & 31;
  const int wid  = tid >> 5;
  const int wm   = (wid & 3) * 32;
  const int wn   = (wid >> 2) * 64;
  const int ln   = lane & 15;
  const int lh   = lane >> 4;

  v8f acc[2][4] = {};

  // tile move mapping: each thread moves 32B for A and 32B for B per stage
  const int r    = tid >> 1;         // 0..127 (row for A, col for B)
  const int half = tid & 1;          // which 16-dword half of the 32-dword row

  const unsigned short* aptr = Ae + (size_t)(m0 + r) * KTOT + half * 32;
  const unsigned*       bptr = Be + (size_t)(n0 + r) * (KTOT / 2) + half * 16;

  auto compute = [&](int buf) {
    #pragma unroll
    for (int s = 0; s < 2; ++s) {        // two K=32 sub-steps per stage
      #pragma unroll
      for (int i = 0; i < 2; ++i) {
        FragU a;
        const int row = wm + 16 * i + ln;
        a.i[0] = *(const int4*)&As[buf][row][s * 16 + 4 * lh];
        a.i[1] = *(const int4*)&As[buf][row][s * 16 + 8 + 4 * lh];
        #pragma unroll
        for (int j = 0; j < 4; ++j) {
          FragU b;
          const int col = wn + 16 * j + ln;
          b.i[0] = *(const int4*)&Bs[buf][col][s * 16 + 4 * lh];
          b.i[1] = *(const int4*)&Bs[buf][col][s * 16 + 8 + 4 * lh];
          acc[i][j] = __builtin_amdgcn_wmma_f32_16x16x32_bf16(
              false, a.v, false, b.v, (short)0, acc[i][j], false, false);
        }
      }
    }
  };

  const int KT = KTOT / BK;

#if USE_ASYNC_LDS
  // ---- async global->LDS DMA path: no VGPR staging, ASYNCcnt-tracked ----
  AS3 char* sA = (AS3 char*)&As[0][0][0];
  AS3 char* sB = (AS3 char*)&Bs[0][0][0];
  const unsigned aoff = (unsigned)(r * LDST + half * 16) * 4u;   // byte offset in one buffer
  const unsigned boff = aoff;
  const unsigned bufBytes = (unsigned)(BM * LDST * 4);

  auto issue = [&](int kk, int buf) {
    AS1 vi4* ga = (AS1 vi4*)((const AS1 char*)(const char*)aptr + (size_t)kk * (BK * 2));
    AS1 vi4* gb = (AS1 vi4*)((const AS1 char*)(const char*)bptr + (size_t)kk * (BK * 2));
    AS3 vi4* la = (AS3 vi4*)(sA + buf * bufBytes + aoff);
    AS3 vi4* lb = (AS3 vi4*)(sB + buf * bufBytes + boff);
    __builtin_amdgcn_global_load_async_to_lds_b128(ga, la, 0, 0);
    __builtin_amdgcn_global_load_async_to_lds_b128(ga, la, 16, 0);
    __builtin_amdgcn_global_load_async_to_lds_b128(gb, lb, 0, 0);
    __builtin_amdgcn_global_load_async_to_lds_b128(gb, lb, 16, 0);
    __builtin_prefetch((const char*)aptr + (size_t)(kk + 2) * (BK * 2), 0, 3);
    __builtin_prefetch((const char*)bptr + (size_t)(kk + 2) * (BK * 2), 0, 3);
  };

  issue(0, 0);
  for (int kk = 0; kk < KT; ++kk) {
    const int cur = kk & 1;
    if (kk + 1 < KT) {
      issue(kk + 1, cur ^ 1);
      wait_asynccnt<4>();      // stage kk's 4 DMAs retired (in-order)
    } else {
      wait_asynccnt<0>();
    }
    __syncthreads();           // all waves' DMAs for `cur` landed
    compute(cur);
    __syncthreads();           // done reading before next overwrite
  }
#else
  // ---- fallback: register-staged double buffering ----
  int4 areg[2], breg[2];
  auto g_load = [&](int kk) {
    const int4* ag = (const int4*)(aptr + kk * BK);
    areg[0] = ag[0]; areg[1] = ag[1];
    const int4* bg = (const int4*)(bptr + kk * (BK / 2));
    breg[0] = bg[0]; breg[1] = bg[1];
    __builtin_prefetch((const char*)(aptr + (kk + 2) * BK), 0, 3);
    __builtin_prefetch((const char*)(bptr + (kk + 2) * (BK / 2)), 0, 3);
  };
  auto s_store = [&](int buf) {
    int4* ad = (int4*)&As[buf][r][half * 16];
    ad[0] = areg[0]; ad[1] = areg[1];
    int4* bd = (int4*)&Bs[buf][r][half * 16];
    bd[0] = breg[0]; bd[1] = breg[1];
  };
  g_load(0); s_store(0); __syncthreads();
  for (int kk = 0; kk < KT; ++kk) {
    const int cur = kk & 1;
    if (kk + 1 < KT) g_load(kk + 1);
    compute(cur);
    if (kk + 1 < KT) s_store(cur ^ 1);
    __syncthreads();
  }
#endif

  // epilogue
  if (DO_GELU) {
    unsigned short* Ce = (unsigned short*)Cout + (size_t)e * TOK * NOUT;
    #pragma unroll
    for (int i = 0; i < 2; ++i) {
      #pragma unroll
      for (int j = 0; j < 4; ++j) {
        const int col  = n0 + wn + 16 * j + ln;
        const int rowb = m0 + wm + 16 * i + 8 * lh;
        #pragma unroll
        for (int rr = 0; rr < 8; ++rr) {
          float x = acc[i][j][rr];
          float g = 0.5f * x * (1.0f + erff(x * 0.70710678118654752f));
          Ce[(size_t)(rowb + rr) * NOUT + col] = bf1(g);
        }
      }
    }
  } else {
    float* Ce = (float*)Cout + (size_t)e * TOK * NOUT;
    #pragma unroll
    for (int i = 0; i < 2; ++i) {
      #pragma unroll
      for (int j = 0; j < 4; ++j) {
        const int col  = n0 + wn + 16 * j + ln;
        const int rowb = m0 + wm + 16 * i + 8 * lh;
        #pragma unroll
        for (int rr = 0; rr < 8; ++rr) {
          Ce[(size_t)(rowb + rr) * NOUT + col] = acc[i][j][rr];
        }
      }
    }
  }
}

// ---------------------------------------------------------------------------
extern "C" void kernel_launch(void* const* d_in, const int* in_sizes, int n_in,
                              void* d_out, int out_size, void* d_ws, size_t ws_size,
                              hipStream_t stream) {
  const float* x  = (const float*)d_in[0];   // [E, TOK, HID]
  const float* w1 = (const float*)d_in[1];   // [E, HID, INTR]
  const float* w2 = (const float*)d_in[2];   // [E, INTR, HID]
  float* out = (float*)d_out;                // [E, TOK, HID]

  // workspace layout:
  //   Xb  @ 0      (32 MB)   W1P @ 32 MB  (64 MB)
  //   W2P @ 96 MB  (64 MB)   Hb  @ 160 MB (128 MB)
  char* ws = (char*)d_ws;
  unsigned*       Xb  = (unsigned*)(ws);
  unsigned*       W1P = (unsigned*)(ws + (size_t)32 * 1024 * 1024);
  unsigned*       W2P = (unsigned*)(ws + (size_t)96 * 1024 * 1024);
  unsigned short* Hb  = (unsigned short*)(ws + (size_t)160 * 1024 * 1024);

  dim3 blk(256);

  {
    const size_t nelem = (size_t)NEXP * TOK * HID;
    convert_x_kernel<<<dim3(nelem / (256 * 8)), blk, 0, stream>>>(x, Xb);
  }
  convert_transpose_w_kernel<<<dim3(INTR / 64, HID / 64, NEXP), blk, 0, stream>>>(
      w1, W1P, HID, INTR);
  convert_transpose_w_kernel<<<dim3(HID / 64, INTR / 64, NEXP), blk, 0, stream>>>(
      w2, W2P, INTR, HID);

  gemm_bf16_kernel<HID, INTR, true><<<dim3(INTR / BN, TOK / BM, NEXP), blk, 0, stream>>>(
      (const unsigned short*)Xb, W1P, (void*)Hb);
  gemm_bf16_kernel<INTR, HID, false><<<dim3(HID / BN, TOK / BM, NEXP), blk, 0, stream>>>(
      Hb, W2P, (void*)out);
}
